// FusedWindowTransformer_66322884985428
// MI455X (gfx1250) — compile-verified, matrix operand order checked by reading, not run on hardware
//
#include <hip/hip_runtime.h>
#include <hip/hip_bf16.h>

// ---------------------------------------------------------------------------
// FusedWindowTransformer for MI455X (gfx1250): bf16 WMMA GEMM pipeline with
// double-buffered async global->LDS staging (ASYNCcnt pipelining) and
// LDS-staged bias/masks in attention.
// B=8 NW=64 WS=64 RS=96 SS=32 DIM=512 NH=8 HD=64 MULT=4 MC=1 T=2080 MAXD=80
// ---------------------------------------------------------------------------

typedef __bf16 bf16;
typedef __bf16 v16bf __attribute__((ext_vector_type(16)));
typedef float  v8f   __attribute__((ext_vector_type(8)));

#define B_    8
#define NW_   64
#define WS_   64
#define RS_   96
#define SS_   32
#define DIM_  512
#define NH_   8
#define HD_   64
#define T_    2080
#define SCALE_ 0.125f

// ----------------------------- workspace layout ----------------------------
constexpr size_t SZ_XN    = (size_t)32768 * 512 * 2;
constexpr size_t OFF_XN   = 0;                                // xn (reused as attn O)
constexpr size_t OFF_XN2  = OFF_XN + SZ_XN;                   // xn_ ; reused for oproj f32
constexpr size_t SZ_XN2R  = (size_t)32768 * 512 * 4;
constexpr size_t OFF_Q    = OFF_XN2 + SZ_XN2R;
constexpr size_t SZ_Q     = (size_t)32768 * 512 * 2;
constexpr size_t OFF_KV   = OFF_Q + SZ_Q;
constexpr size_t SZ_KV    = (size_t)49152 * 1024 * 2;
constexpr size_t OFF_Y    = OFF_KV + SZ_KV;
constexpr size_t SZ_Y     = (size_t)16640 * 512 * 4;
constexpr size_t OFF_YN   = OFF_Y + SZ_Y;
constexpr size_t SZ_YN    = (size_t)16640 * 512 * 2;
constexpr size_t OFF_WQT  = OFF_YN + SZ_YN;
constexpr size_t OFF_WKVT = OFF_WQT  + (size_t)512 * 512 * 2;
constexpr size_t OFF_WPT  = OFF_WKVT + (size_t)1024 * 512 * 2;
constexpr size_t OFF_WF1T = OFF_WPT  + (size_t)512 * 512 * 2;
constexpr size_t OFF_WF2T = OFF_WF1T + (size_t)2048 * 512 * 2;
constexpr size_t OFF_O    = OFF_XN;
constexpr size_t OFF_OPROJ= OFF_XN2;
constexpr size_t OFF_H1   = OFF_Q;

// ----------------------------- CDNA5 helpers -------------------------------
union Frag16 { v16bf v; unsigned int u[8]; };
union U4H    { uint4 u; bf16 e[8]; };

// 16x32 bf16 fragment from a row-major tile, matching the CDNA5 16-bit
// operand layout: lane l<16: row0+l, K=col0+{0..7,16..23}; lane l>=16:
// row0+l-16, K=col0+{8..15,24..31}.
__device__ __forceinline__ v16bf ldfrag(const bf16* base, int ld, int row0, int col0, int lane) {
  int l   = lane & 15;
  int klo = (lane & 16) ? 8 : 0;
  const bf16* p = base + (size_t)(row0 + l) * ld + col0 + klo;
  Frag16 f;
#pragma unroll
  for (int i = 0; i < 4; ++i) {
    f.u[i]     = *(const unsigned int*)(p + 2 * i);
    f.u[4 + i] = *(const unsigned int*)(p + 16 + 2 * i);
  }
  return f.v;
}

__device__ __forceinline__ v8f wmma_bf16(v16bf a, v16bf b, v8f c) {
  return __builtin_amdgcn_wmma_f32_16x16x32_bf16(false, a, false, b, (short)0, c, false, false);
}

__device__ __forceinline__ v8f zero8() {
  v8f z = {0.f, 0.f, 0.f, 0.f, 0.f, 0.f, 0.f, 0.f};
  return z;
}

// Async global->LDS copy, 16B per lane (GLOBAL_LOAD_ASYNC_TO_LDS_B128,
// tracked by ASYNCcnt). LDS byte address = low 32 bits of generic pointer
// (ISA 10.2).
__device__ __forceinline__ void async_copy_b128(void* lds_dst, const void* gsrc) {
  unsigned lds_addr = (unsigned)(unsigned long long)lds_dst;
  asm volatile("global_load_async_to_lds_b128 %0, %1, off"
               :: "v"(lds_addr), "v"(gsrc)
               : "memory");
}
template <int N>
__device__ __forceinline__ void wait_async() {
  asm volatile("s_wait_asynccnt %0" :: "i"(N) : "memory");
}

// ----------------------------- LayerNorm (row of 512) ----------------------
__global__ void ln_rows_kernel(const float* __restrict__ in, const float* __restrict__ g,
                               const float* __restrict__ bb, bf16* __restrict__ out) {
  __shared__ float red[8];
  __shared__ float stat[2];
  int row = blockIdx.x, tid = threadIdx.x;
  const float* p = in + (size_t)row * DIM_;
  float v0 = p[tid], v1 = p[tid + 256];
  float s = v0 + v1;
#pragma unroll
  for (int o = 16; o > 0; o >>= 1) s += __shfl_xor(s, o, 32);
  if ((tid & 31) == 0) red[tid >> 5] = s;
  __syncthreads();
  if (tid == 0) {
    float t = 0.f;
    for (int i = 0; i < 8; ++i) t += red[i];
    stat[0] = t * (1.0f / DIM_);
  }
  __syncthreads();
  float mean = stat[0];
  float d0 = v0 - mean, d1 = v1 - mean;
  float q = d0 * d0 + d1 * d1;
#pragma unroll
  for (int o = 16; o > 0; o >>= 1) q += __shfl_xor(q, o, 32);
  if ((tid & 31) == 0) red[tid >> 5] = q;
  __syncthreads();
  if (tid == 0) {
    float t = 0.f;
    for (int i = 0; i < 8; ++i) t += red[i];
    stat[1] = rsqrtf(t * (1.0f / DIM_) + 1e-5f);
  }
  __syncthreads();
  float rstd = stat[1];
  bf16* po = out + (size_t)row * DIM_;
  po[tid]       = (bf16)(d0 * rstd * g[tid] + bb[tid]);
  po[tid + 256] = (bf16)(d1 * rstd * g[tid + 256] + bb[tid + 256]);
}

// ------------------- weight transpose+convert: wt[n][k] = w[k][n] ----------
__global__ void wconv_kernel(const float* __restrict__ w, bf16* __restrict__ wt,
                             int K, int N) {
  int idx = blockIdx.x * 256 + threadIdx.x;
  if (idx >= N * K) return;
  int n = idx / K, k = idx - n * K;
  wt[idx] = (bf16)w[(size_t)k * N + n];
}

// ----------------------------- WMMA GEMM -----------------------------------
// C[M,N] = A[M,K](bf16 rm) @ BT[N,K](bf16)^T + bias, epilogue EPI:
//   0: f32; 1: bf16; 2: GELU(exact)->bf16; 3: +res -> f32
// Block tile 128x128, 256 threads (8 waves), wave tile 32x64 (2x4 WMMA).
// Double-buffered async global->LDS staging: issue tile k+1 (4 ops/wave),
// s_wait_asynccnt 4 retires tile k, compute overlaps copy of k+1.
// Requires M%128==0, N%128==0, K%32==0 (true at all call sites).
#define SA_LD 40
template <int EPI>
__global__ void gemm_bf16_wmma(const bf16* __restrict__ A, const bf16* __restrict__ BT,
                               const float* __restrict__ bias, const float* __restrict__ res,
                               float* __restrict__ outF, bf16* __restrict__ outB,
                               int M, int N, int K) {
  __shared__ bf16 sA[2][128 * SA_LD];
  __shared__ bf16 sB[2][128 * SA_LD];
  int tid = threadIdx.x, wv = tid >> 5, lane = tid & 31;
  int m0 = blockIdx.x * 128, n0 = blockIdx.y * 128;
  int wm = (wv & 3) * 32, wn = (wv >> 2) * 64;

  v8f acc[2][4];
#pragma unroll
  for (int i = 0; i < 2; ++i)
#pragma unroll
    for (int j = 0; j < 4; ++j) acc[i][j] = zero8();

  // per-thread staging coordinates: 128x32 tile = 512 uint4, 2 per thread
  int r0 = tid >> 2, c0 = (tid & 3) * 8;
  int r1 = r0 + 64, c1 = c0;
  const bf16* aSrc0 = A  + (size_t)(m0 + r0) * K + c0;
  const bf16* aSrc1 = A  + (size_t)(m0 + r1) * K + c1;
  const bf16* bSrc0 = BT + (size_t)(n0 + r0) * K + c0;
  const bf16* bSrc1 = BT + (size_t)(n0 + r1) * K + c1;

  auto issue = [&](int p, int koff) {
    async_copy_b128(&sA[p][r0 * SA_LD + c0], aSrc0 + koff);
    async_copy_b128(&sA[p][r1 * SA_LD + c1], aSrc1 + koff);
    async_copy_b128(&sB[p][r0 * SA_LD + c0], bSrc0 + koff);
    async_copy_b128(&sB[p][r1 * SA_LD + c1], bSrc1 + koff);
  };

  int nk = K >> 5;
  issue(0, 0);
  for (int k = 0; k < nk; ++k) {
    int p = k & 1;
    if (k + 1 < nk) {
      issue(1 - p, (k + 1) * 32);  // prefetch next tile into other buffer
      wait_async<4>();             // retire tile k, leave tile k+1 in flight
    } else {
      wait_async<0>();
    }
    __syncthreads();

    v16bf a0 = ldfrag(sA[p], SA_LD, wm + 0, 0, lane);
    v16bf a1 = ldfrag(sA[p], SA_LD, wm + 16, 0, lane);
    v16bf b0 = ldfrag(sB[p], SA_LD, wn + 0, 0, lane);
    v16bf b1 = ldfrag(sB[p], SA_LD, wn + 16, 0, lane);
    v16bf b2 = ldfrag(sB[p], SA_LD, wn + 32, 0, lane);
    v16bf b3 = ldfrag(sB[p], SA_LD, wn + 48, 0, lane);
    acc[0][0] = wmma_bf16(a0, b0, acc[0][0]);
    acc[0][1] = wmma_bf16(a0, b1, acc[0][1]);
    acc[0][2] = wmma_bf16(a0, b2, acc[0][2]);
    acc[0][3] = wmma_bf16(a0, b3, acc[0][3]);
    acc[1][0] = wmma_bf16(a1, b0, acc[1][0]);
    acc[1][1] = wmma_bf16(a1, b1, acc[1][1]);
    acc[1][2] = wmma_bf16(a1, b2, acc[1][2]);
    acc[1][3] = wmma_bf16(a1, b3, acc[1][3]);
    __syncthreads();  // all waves done reading buf p before its next overwrite
  }

  int rbase = (lane & 16) ? 8 : 0;
  int col = lane & 15;
#pragma unroll
  for (int i = 0; i < 2; ++i) {
#pragma unroll
    for (int j = 0; j < 4; ++j) {
      int gcol = n0 + wn + j * 16 + col;
      float bv = bias[gcol];
#pragma unroll
      for (int r = 0; r < 8; ++r) {
        int grow = m0 + wm + i * 16 + rbase + r;
        float v = acc[i][j][r] + bv;
        size_t o = (size_t)grow * N + gcol;
        if constexpr (EPI == 0) {
          outF[o] = v;
        } else if constexpr (EPI == 1) {
          outB[o] = (bf16)v;
        } else if constexpr (EPI == 2) {
          outB[o] = (bf16)(0.5f * v * (1.0f + erff(v * 0.70710678118654752f)));
        } else {
          outF[o] = v + res[o];
        }
      }
    }
  }
}

// ----------------------------- Attention -----------------------------------
// One block per (window w in [0,512), head h in [0,8)). 256 threads, 8 waves.
#define SQ_LD  72
#define SK_LD  72
#define SVT_LD 104
#define SSC_LD 100
#define SP_LD  104
#define SMEM_PH1   (64 * SQ_LD * 2 + 96 * SK_LD * 2)             // 23040
#define SMEM_BYTES (SMEM_PH1 + 64 * SSC_LD * 4 + 64 * SP_LD * 2) // 61952

__global__ void attn_kernel(const bf16* __restrict__ Q, const bf16* __restrict__ KV,
                            const float* __restrict__ bias_table,
                            const int* __restrict__ mask_left,
                            const int* __restrict__ mask_right,
                            bf16* __restrict__ O) {
  __shared__ __align__(16) unsigned char smem[SMEM_BYTES];
  __shared__ float rpbS[160];                           // bias_table slice for head h
  bf16* sQ  = (bf16*)smem;                              // phase 1
  bf16* sK  = (bf16*)(smem + 64 * SQ_LD * 2);           // phase 1
  bf16* sVT = (bf16*)smem;                              // phase 2 (overlaps sQ/sK)
  float* sS = (float*)(smem + SMEM_PH1);
  bf16* sP  = (bf16*)(smem + SMEM_PH1 + 64 * SSC_LD * 4);
  unsigned char* sM = (unsigned char*)sP;               // mask bytes, consumed pre-softmax

  int w = blockIdx.x;            // global window
  int h = blockIdx.y;            // head
  int win = w & (NW_ - 1);       // window index within batch
  int tid = threadIdx.x, wv = tid >> 5, lane = tid & 31;

  // stage Q[64][64] head slice: 2 async b128 per thread
  {
    int i = tid >> 2, cq = (tid & 3) * 16;
    const bf16* src = Q + ((size_t)w * WS_ + i) * (NH_ * HD_) + h * HD_ + cq;
    async_copy_b128(sQ + i * SQ_LD + cq, src);
    async_copy_b128(sQ + i * SQ_LD + cq + 8, src + 8);
  }
  // stage K[96][64] head slice: 768 x 16B, 3 per thread
#pragma unroll
  for (int s = 0; s < 3; ++s) {
    int u = tid + s * 256;
    int r = u >> 3;
    int ck = (u & 7) * 8;
    async_copy_b128(sK + r * SK_LD + ck,
                    KV + ((size_t)w * RS_ + r) * (2 * NH_ * HD_) + h * HD_ + ck);
  }
  // stage rpb slice for this head
  if (tid < 160) rpbS[tid] = bias_table[(size_t)tid * NH_ + h];
  // stage boundary mask bytes (uniform per block); interior windows skip
  const int* mask = (win == 0) ? mask_left : ((win == NW_ - 1) ? mask_right : nullptr);
  if (mask) {
#pragma unroll
    for (int s = 0; s < 24; ++s) {
      int e = tid + s * 256;  // 64*96
      sM[e] = (unsigned char)mask[e];
    }
  }
  int mact = mask ? 0xFF : 0;  // branchless mask activity flag
  wait_async<0>();
  __syncthreads();

  // scores = Q K^T : 4x6=24 tiles of 16x16, 3 per wave; contraction HD=64
#pragma unroll
  for (int s = 0; s < 3; ++s) {
    int t = wv * 3 + s;
    int mt = t / 6, nt = t % 6;
    v8f acc = zero8();
#pragma unroll
    for (int ks = 0; ks < HD_; ks += 32) {
      v16bf a = ldfrag(sQ, SQ_LD, mt * 16, ks, lane);
      v16bf b = ldfrag(sK, SK_LD, nt * 16, ks, lane);
      acc = wmma_bf16(a, b, acc);
    }
    int rbase = (lane & 16) ? 8 : 0;
    int j = nt * 16 + (lane & 15);
#pragma unroll
    for (int r = 0; r < 8; ++r) {
      int i = mt * 16 + rbase + r;
      float v = acc[r] * SCALE_ + rpbS[i - j + 96];
      v = (sM[i * RS_ + j] & mact) ? -3.0e38f : v;  // v_cndmask, no branch
      sS[i * SSC_LD + j] = v;
    }
  }
  __syncthreads();

  // phase 2: stage V^T[64][96] (overlaps sQ/sK) while 64 threads run softmax.
  // Read V rows as uint4 (16B), scatter 8 ds_store_b16 each (LDS transpose).
#pragma unroll
  for (int s = 0; s < 3; ++s) {
    int e = tid + s * 256;      // 768 chunks of 8 elements
    int kr = e >> 3;
    int nc = (e & 7) * 8;
    U4H vv;
    vv.u = *(const uint4*)(KV + ((size_t)w * RS_ + kr) * (2 * NH_ * HD_) + NH_ * HD_ + h * HD_ + nc);
#pragma unroll
    for (int q = 0; q < 8; ++q) sVT[(nc + q) * SVT_LD + kr] = vv.e[q];
  }
  if (tid < WS_) {
    float mx = -3.4e38f;
    for (int j = 0; j < RS_; ++j) mx = fmaxf(mx, sS[tid * SSC_LD + j]);
    float sum = 0.f;
    for (int j = 0; j < RS_; ++j) {
      float ev = __expf(sS[tid * SSC_LD + j] - mx);
      sum += ev;
      sS[tid * SSC_LD + j] = ev;
    }
    float inv = 1.0f / sum;
    for (int j = 0; j < RS_; ++j) sP[tid * SP_LD + j] = (bf16)(sS[tid * SSC_LD + j] * inv);
  }
  __syncthreads();

  // O = P V : 4x4=16 tiles, 2 per wave; contraction RS=96
#pragma unroll
  for (int s = 0; s < 2; ++s) {
    int t = wv * 2 + s;
    int mt = t >> 2, nt = t & 3;
    v8f acc = zero8();
#pragma unroll
    for (int ks = 0; ks < RS_; ks += 32) {
      v16bf a = ldfrag(sP, SP_LD, mt * 16, ks, lane);
      v16bf b = ldfrag(sVT, SVT_LD, nt * 16, ks, lane);
      acc = wmma_bf16(a, b, acc);
    }
    int rbase = (lane & 16) ? 8 : 0;
    int col = lane & 15;
#pragma unroll
    for (int r = 0; r < 8; ++r) {
      int i = mt * 16 + rbase + r;
      int n = nt * 16 + col;
      O[((size_t)w * WS_ + i) * (NH_ * HD_) + h * HD_ + n] = (bf16)acc[r];
    }
  }
}

// ------------- gather overlapping windows + count-normalize + residual -----
__global__ void gather_kernel(const float* __restrict__ oproj, const float* __restrict__ x,
                              float* __restrict__ y) {
  size_t idx = (size_t)blockIdx.x * 256 + threadIdx.x;
  if (idx >= (size_t)B_ * T_ * DIM_) return;
  int d = (int)(idx & (DIM_ - 1));
  int t = (int)((idx >> 9) % T_);
  int b = (int)(idx / ((size_t)DIM_ * T_));
  int wlo = (t >= SS_) ? ((t - SS_) >> 5) : 0;
  int whi = t >> 5; if (whi > NW_ - 1) whi = NW_ - 1;
  float s = 0.f;
  for (int w = wlo; w <= whi; ++w)
    s += oproj[((size_t)(b * NW_ + w) * WS_ + (t - w * SS_)) * DIM_ + d];
  y[idx] = s / (float)(whi - wlo + 1) + x[idx];
}

// ---------------------------------------------------------------------------
extern "C" void kernel_launch(void* const* d_in, const int* in_sizes, int n_in,
                              void* d_out, int out_size, void* d_ws, size_t ws_size,
                              hipStream_t stream) {
  const float* x          = (const float*)d_in[0];
  const float* windowX    = (const float*)d_in[1];
  const float* windowX_   = (const float*)d_in[2];
  const int*   mask_left  = (const int*)d_in[3];
  const int*   mask_right = (const int*)d_in[4];
  // d_in[5] = nW (compile-time constant 64)
  const float* b_q        = (const float*)d_in[7];
  const float* b_kv       = (const float*)d_in[9];
  const float* b_proj     = (const float*)d_in[11];
  const float* bias_table = (const float*)d_in[12];
  const float* ln_attn_g  = (const float*)d_in[13];
  const float* ln_attn_b  = (const float*)d_in[14];
  const float* ln_mlp_g   = (const float*)d_in[15];
  const float* ln_mlp_b   = (const float*)d_in[16];
  const float* b_fc1      = (const float*)d_in[18];
  const float* b_fc2      = (const float*)d_in[20];
  float* out = (float*)d_out;
  char*  ws  = (char*)d_ws;

  bf16* xn    = (bf16*)(ws + OFF_XN);
  bf16* xn_   = (bf16*)(ws + OFF_XN2);
  bf16* Qbf   = (bf16*)(ws + OFF_Q);
  bf16* KVbf  = (bf16*)(ws + OFF_KV);
  bf16* Obf   = (bf16*)(ws + OFF_O);
  float* oprj = (float*)(ws + OFF_OPROJ);
  float* y    = (float*)(ws + OFF_Y);
  bf16* yn    = (bf16*)(ws + OFF_YN);
  bf16* h1    = (bf16*)(ws + OFF_H1);
  bf16* wqT   = (bf16*)(ws + OFF_WQT);
  bf16* wkvT  = (bf16*)(ws + OFF_WKVT);
  bf16* wpT   = (bf16*)(ws + OFF_WPT);
  bf16* wf1T  = (bf16*)(ws + OFF_WF1T);
  bf16* wf2T  = (bf16*)(ws + OFF_WF2T);

  // 1) LayerNorms -> bf16
  ln_rows_kernel<<<32768, 256, 0, stream>>>(windowX,  ln_attn_g, ln_attn_b, xn);
  ln_rows_kernel<<<49152, 256, 0, stream>>>(windowX_, ln_attn_g, ln_attn_b, xn_);

  // 2) weights -> bf16 [N][K]
  wconv_kernel<<<(512 * 512 + 255) / 256, 256, 0, stream>>>((const float*)d_in[6], wqT, 512, 512);
  wconv_kernel<<<(512 * 1024 + 255) / 256, 256, 0, stream>>>((const float*)d_in[8], wkvT, 512, 1024);
  wconv_kernel<<<(512 * 512 + 255) / 256, 256, 0, stream>>>((const float*)d_in[10], wpT, 512, 512);
  wconv_kernel<<<(512 * 2048 + 255) / 256, 256, 0, stream>>>((const float*)d_in[17], wf1T, 512, 2048);
  wconv_kernel<<<(2048 * 512 + 255) / 256, 256, 0, stream>>>((const float*)d_in[19], wf2T, 2048, 512);

  // 3) Q = xn @ w_q + b_q  -> bf16 [32768,512]
  gemm_bf16_wmma<1><<<dim3(32768 / 128, 512 / 128), 256, 0, stream>>>(
      xn, wqT, b_q, nullptr, nullptr, Qbf, 32768, 512, 512);
  // 4) KV = xn_ @ w_kv + b_kv -> bf16 [49152,1024]
  gemm_bf16_wmma<1><<<dim3(49152 / 128, 1024 / 128), 256, 0, stream>>>(
      xn_, wkvT, b_kv, nullptr, nullptr, KVbf, 49152, 1024, 512);

  // 5) windowed attention -> O bf16 [32768,512]
  attn_kernel<<<dim3(B_ * NW_, NH_), 256, 0, stream>>>(Qbf, KVbf, bias_table,
                                                       mask_left, mask_right, Obf);

  // 6) out projection -> f32 [32768,512]
  gemm_bf16_wmma<0><<<dim3(32768 / 128, 512 / 128), 256, 0, stream>>>(
      Obf, wpT, b_proj, nullptr, oprj, nullptr, 32768, 512, 512);

  // 7) overlap-gather + normalize + residual -> y f32 [B,T,DIM]
  gather_kernel<<<(int)(((size_t)B_ * T_ * DIM_ + 255) / 256), 256, 0, stream>>>(oprj, x, y);

  // 8) MLP: LN -> fc1+GELU -> fc2 + residual -> d_out
  ln_rows_kernel<<<16640, 256, 0, stream>>>(y, ln_mlp_g, ln_mlp_b, yn);
  gemm_bf16_wmma<2><<<dim3(16640 / 128, 2048 / 128), 256, 0, stream>>>(
      yn, wf1T, b_fc1, nullptr, nullptr, h1, 16640, 2048, 512);
  gemm_bf16_wmma<3><<<dim3(16640 / 128, 512 / 128), 256, 0, stream>>>(
      h1, wf2T, b_fc2, y, out, nullptr, 16640, 512, 2048);
}
